// SparseGATLayer_31250182046216
// MI455X (gfx1250) — matrix operand order repaired and checked by reading, not compiled, not run on hardware
//
#include <hip/hip_runtime.h>
#include <hip/hip_bf16.h>
#include <string.h>

typedef __attribute__((ext_vector_type(16))) _Float16 v16h;
typedef __attribute__((ext_vector_type(8)))  float    v8f;

#define NN 100000
#define EE 640000
#define HH 4
#define DD 32
#define FDIM 128
#define NEG_SLOPE 0.2f

// ---------- order-preserving float<->int map (involution) ----------
__device__ __forceinline__ int f2ord(float f) {
    int i = __float_as_int(f);
    return i >= 0 ? i : (i ^ 0x7FFFFFFF);
}
__device__ __forceinline__ float ord2f(int i) {
    return __int_as_float(i >= 0 ? i : (i ^ 0x7FFFFFFF));
}

// ---------- fill ----------
__global__ void fill_u32(unsigned int* p, unsigned int v, int n) {
    int i = blockIdx.x * blockDim.x + threadIdx.x;
    if (i < n) p[i] = v;
}

// ---------- fused score weights: M20[128][20] ----------
// j in 0..3  : Aq[i][h]      = sum_d Wq[i][h*32+d] * attn[h*64+d]
// j in 4..7  : Ak[i][h]      = sum_d Wk[i][h*32+d] * attn[h*64+32+d]
// j in 8..19 : Ase[i][t*4+h] = sum_d Wq[i][h*32+d] * ee[t*128+h*32+d]
__global__ void fuse_weights(const float* __restrict__ Wq, const float* __restrict__ Wk,
                             const float* __restrict__ attn, const float* __restrict__ ee,
                             float* __restrict__ M20) {
    int j = blockIdx.x;       // 0..19
    int i = threadIdx.x;      // 0..127
    float acc = 0.f;
    if (j < 4) {
        int h = j;
        for (int d = 0; d < DD; ++d)
            acc += Wq[i * FDIM + h * DD + d] * attn[h * 64 + d];
    } else if (j < 8) {
        int h = j - 4;
        for (int d = 0; d < DD; ++d)
            acc += Wk[i * FDIM + h * DD + d] * attn[h * 64 + 32 + d];
    } else {
        int jj = j - 8, t = jj >> 2, h = jj & 3;
        for (int d = 0; d < DD; ++d)
            acc += Wq[i * FDIM + h * DD + d] * ee[t * FDIM + h * DD + d];
    }
    M20[i * 20 + j] = acc;
}

// ---------- transpose + cvt weights to f16: WT[n][k] = W[k][n] ----------
__global__ void transpose_cvt(const float* __restrict__ W, _Float16* __restrict__ WT) {
    int k = blockIdx.x;   // 0..127
    int n = threadIdx.x;  // 0..127
    WT[n * FDIM + k] = (_Float16)W[k * FDIM + n];
}

// ---------- per-node score scalars: S[n][20] = x[n,:] @ M20 ----------
__global__ void node_scalars(const float* __restrict__ x, const float* __restrict__ M20,
                             float* __restrict__ S) {
    int n = blockIdx.x;
    int j = threadIdx.x;  // 32 threads, 20 active
    if (j >= 20) return;
    const float* xr = x + (size_t)n * FDIM;
    float acc = 0.f;
    for (int i = 0; i < FDIM; ++i)
        acc += xr[i] * M20[i * 20 + j];
    S[(size_t)n * 20 + j] = acc;
}

// ---------- edge pass A: scores + segment max ----------
__global__ void edge_pass_a(const float* __restrict__ S, const int* __restrict__ src,
                            const int* __restrict__ dst, const int* __restrict__ etype,
                            float4* __restrict__ scores, int* __restrict__ m, int E) {
    int e = blockIdx.x * blockDim.x + threadIdx.x;
    if (e >= E) return;
    int s = src[e], d = dst[e], t = etype[e];
    float4 vq = *(const float4*)(S + (size_t)s * 20);
    float4 vk = *(const float4*)(S + (size_t)d * 20 + 4);
    float4 ve = *(const float4*)(S + (size_t)s * 20 + 8 + t * 4);
    float z0 = vq.x + vk.x; z0 = z0 >= 0.f ? z0 : NEG_SLOPE * z0; z0 += ve.x;
    float z1 = vq.y + vk.y; z1 = z1 >= 0.f ? z1 : NEG_SLOPE * z1; z1 += ve.y;
    float z2 = vq.z + vk.z; z2 = z2 >= 0.f ? z2 : NEG_SLOPE * z2; z2 += ve.z;
    float z3 = vq.w + vk.w; z3 = z3 >= 0.f ? z3 : NEG_SLOPE * z3; z3 += ve.w;
    scores[e] = make_float4(z0, z1, z2, z3);
    atomicMax(&m[d * 4 + 0], f2ord(z0));
    atomicMax(&m[d * 4 + 1], f2ord(z1));
    atomicMax(&m[d * 4 + 2], f2ord(z2));
    atomicMax(&m[d * 4 + 3], f2ord(z3));
}

// ---------- edge pass B: ex, denom, unnormalized message scatter (1 wave/edge) ----------
__global__ void edge_pass_b(const float4* __restrict__ scores, const int* __restrict__ m,
                            const int* __restrict__ src, const int* __restrict__ dst,
                            const float* __restrict__ v, float* __restrict__ denom,
                            float* __restrict__ out_pre, int E) {
    int wave = blockIdx.x * (blockDim.x >> 5) + (threadIdx.x >> 5);
    if (wave >= E) return;
    int lane = threadIdx.x & 31;
    int s = src[wave], d = dst[wave];
    float4 sc = scores[wave];
    int4 mo = *(const int4*)(m + (size_t)d * 4);
    float e0 = expf(sc.x - ord2f(mo.x));
    float e1 = expf(sc.y - ord2f(mo.y));
    float e2 = expf(sc.z - ord2f(mo.z));
    float e3 = expf(sc.w - ord2f(mo.w));
    if (lane < 4) {
        float myex = lane == 0 ? e0 : lane == 1 ? e1 : lane == 2 ? e2 : e3;
        atomicAdd(&denom[(size_t)d * 4 + lane], myex);
    }
    const float* vr = v + (size_t)s * FDIM;
    float* orow = out_pre + (size_t)d * FDIM;
    atomicAdd(&orow[0 * 32 + lane], e0 * vr[0 * 32 + lane]);
    atomicAdd(&orow[1 * 32 + lane], e1 * vr[1 * 32 + lane]);
    atomicAdd(&orow[2 * 32 + lane], e2 * vr[2 * 32 + lane]);
    atomicAdd(&orow[3 * 32 + lane], e3 * vr[3 * 32 + lane]);
}

// ---------- WMMA GEMM: C[N x 128] = scale(A) @ BT^T (+bias) ----------
// A f32 row-major, BT f16 [n][k], denomOpt: per (row, kt) softmax scale, bias optional.
// One wave per 16-row strip; 8 col tiles; K loop of 4 (k-step 32).
__global__ void gemm_wmma16(const float* __restrict__ A, const _Float16* __restrict__ BT,
                            const float* __restrict__ denomOpt, const float* __restrict__ bias,
                            float* __restrict__ C, int rowTiles) {
    int wave = blockIdx.x * (blockDim.x >> 5) + (threadIdx.x >> 5);
    if (wave >= rowTiles) return;
    int lane = threadIdx.x & 31;
    int half = lane >> 4;
    int mrow = lane & 15;
    int row0 = wave * 16;

    v8f acc[8];
    for (int ct = 0; ct < 8; ++ct)
        for (int r = 0; r < 8; ++r) acc[ct][r] = 0.f;

    for (int kt = 0; kt < 4; ++kt) {
        // per-fragment softmax normalization scale (whole K-slice of 32 == head kt)
        float sc = 1.f;
        if (denomOpt) sc = 1.f / (denomOpt[(size_t)(row0 + mrow) * 4 + kt] + 1e-10f);

        const float* ap = A + (size_t)(row0 + mrow) * FDIM + kt * 32 + half * 16;
        float4 x0 = *(const float4*)(ap + 0);
        float4 x1 = *(const float4*)(ap + 4);
        float4 x2 = *(const float4*)(ap + 8);
        float4 x3 = *(const float4*)(ap + 12);
        v16h a;
        a[0]  = (_Float16)(x0.x * sc); a[1]  = (_Float16)(x0.y * sc);
        a[2]  = (_Float16)(x0.z * sc); a[3]  = (_Float16)(x0.w * sc);
        a[4]  = (_Float16)(x1.x * sc); a[5]  = (_Float16)(x1.y * sc);
        a[6]  = (_Float16)(x1.z * sc); a[7]  = (_Float16)(x1.w * sc);
        a[8]  = (_Float16)(x2.x * sc); a[9]  = (_Float16)(x2.y * sc);
        a[10] = (_Float16)(x2.z * sc); a[11] = (_Float16)(x2.w * sc);
        a[12] = (_Float16)(x3.x * sc); a[13] = (_Float16)(x3.y * sc);
        a[14] = (_Float16)(x3.z * sc); a[15] = (_Float16)(x3.w * sc);

        for (int ct = 0; ct < 8; ++ct) {
            const _Float16* bp = BT + (size_t)(ct * 16 + mrow) * FDIM + kt * 32 + half * 16;
            v16h b = *(const v16h*)bp;
            acc[ct] = __builtin_amdgcn_wmma_f32_16x16x32_f16(
                false, a, false, b, (short)0, acc[ct], false, false);
        }
    }

    for (int ct = 0; ct < 8; ++ct) {
        int ocol = ct * 16 + mrow;
        float bv = bias ? bias[ocol] : 0.f;
        for (int r = 0; r < 8; ++r) {
            int orow = row0 + r + half * 8;
            C[(size_t)orow * FDIM + ocol] = acc[ct][r] + bv;
        }
    }
}

extern "C" void kernel_launch(void* const* d_in, const int* in_sizes, int n_in,
                              void* d_out, int out_size, void* d_ws, size_t ws_size,
                              hipStream_t stream) {
    const float* x    = (const float*)d_in[0];
    const float* Wq   = (const float*)d_in[1];
    const float* Wk   = (const float*)d_in[2];
    const float* Wv   = (const float*)d_in[3];
    const float* attn = (const float*)d_in[4];
    const float* ee   = (const float*)d_in[5];
    const float* Wout = (const float*)d_in[6];
    const float* bout = (const float*)d_in[7];
    const int*   ei   = (const int*)d_in[8];
    const int*   et   = (const int*)d_in[9];
    const int E = in_sizes[9];
    const int N = in_sizes[0] / FDIM;
    const int* src = ei;
    const int* dst = ei + E;

    // workspace carve (256B aligned)
    char* p = (char*)d_ws;
    auto carve = [&](size_t bytes) -> char* {
        char* r = p;
        p += (bytes + 255) & ~(size_t)255;
        return r;
    };
    float*     M20     = (float*)carve((size_t)FDIM * 20 * 4);
    _Float16*  WvT     = (_Float16*)carve((size_t)FDIM * FDIM * 2);
    _Float16*  WoutT   = (_Float16*)carve((size_t)FDIM * FDIM * 2);
    float*     S       = (float*)carve((size_t)N * 20 * 4);
    float*     V       = (float*)carve((size_t)N * FDIM * 4);
    float4*    scores  = (float4*)carve((size_t)E * 4 * 4);
    int*       m       = (int*)carve((size_t)N * 4 * 4);
    float*     denom   = (float*)carve((size_t)N * 4 * 4);
    float*     out_pre = (float*)carve((size_t)N * FDIM * 4);

    // 1) fused score weights + f16 transposed weights
    fuse_weights<<<20, 128, 0, stream>>>(Wq, Wk, attn, ee, M20);
    transpose_cvt<<<FDIM, FDIM, 0, stream>>>(Wv, WvT);
    transpose_cvt<<<FDIM, FDIM, 0, stream>>>(Wout, WoutT);

    // 2) per-node score scalars (exact f32)
    node_scalars<<<N, 32, 0, stream>>>(x, M20, S);

    // 3) v = x @ Wv  (WMMA f16, f32 accumulate)
    int rowTiles = N / 16;
    int gemmBlocks = (rowTiles + 3) / 4;  // 4 waves/block
    gemm_wmma16<<<gemmBlocks, 128, 0, stream>>>(x, WvT, nullptr, nullptr, V, rowTiles);

    // 4) init accumulators: m = ord(-100.0f) = 0xBD37FFFF, denom = 0, out_pre = 0
    fill_u32<<<(N * 4 + 255) / 256, 256, 0, stream>>>((unsigned int*)m, 0xBD37FFFFu, N * 4);
    fill_u32<<<(N * 4 + 255) / 256, 256, 0, stream>>>((unsigned int*)denom, 0u, N * 4);
    fill_u32<<<(N * FDIM + 255) / 256, 256, 0, stream>>>((unsigned int*)out_pre, 0u, N * FDIM);

    // 5) edge pass A: scores + segment max
    edge_pass_a<<<(E + 255) / 256, 256, 0, stream>>>(S, src, dst, et, scores, m, E);

    // 6) edge pass B: exp / denom / unnormalized message scatter (1 wave per edge)
    edge_pass_b<<<(E + 7) / 8, 256, 0, stream>>>(scores, m, src, dst, V, denom, out_pre, E);

    // 7) out = normalize(out_pre) @ Wout + bout  (normalization fused into A-fragment load)
    gemm_wmma16<<<gemmBlocks, 128, 0, stream>>>(out_pre, WoutT, denom, bout, (float*)d_out,
                                                rowTiles);
}
